// TSGCNet_89223650607920
// MI455X (gfx1250) — matrix operand order repaired and checked by reading, not compiled, not run on hardware
//
#include <hip/hip_runtime.h>

#define NPTS 10000
#define NB   2
#define KNN  16

typedef __attribute__((ext_vector_type(16))) __bf16 v16bf;
typedef __attribute__((ext_vector_type(8)))  float  v8f;
typedef __attribute__((ext_vector_type(2)))  float  v2f;

union FragB { v16bf v; unsigned u[8]; };

enum { ACT_NONE = 0, ACT_RELU = 1, ACT_LRELU = 2 };
enum { BSRC_MAT = 0, BSRC_EDGE = 1, BSRC_ATT = 2 };

#define LROW 40   // padded LDS row (ushorts): 80B keeps b128 alignment, ~2-way banks

#if __has_builtin(__builtin_amdgcn_global_load_async_to_lds_b128) && \
    __has_builtin(__builtin_amdgcn_s_wait_asynccnt)
#define HAVE_ASYNC 1
typedef int v4i_async __attribute__((ext_vector_type(4)));
typedef __attribute__((address_space(1))) v4i_async* gvp_t;   // global v4i*
typedef __attribute__((address_space(3))) v4i_async* svp_t;   // LDS v4i*
#else
#define HAVE_ASYNC 0
#endif

static __device__ inline unsigned short bfbits(float f) {
  unsigned u = __builtin_bit_cast(unsigned, f);
  u = u + 0x7FFFu + ((u >> 16) & 1u);           // round-to-nearest-even
  return (unsigned short)(u >> 16);
}
static __device__ inline unsigned pk2(float a, float b) {
  return (unsigned)bfbits(a) | ((unsigned)bfbits(b) << 16);
}

struct GemmP {
  const float* A; int as_m, as_k, as_b;   // weights (M,K)
  const float* B; int bs_k, bs_n, bs_b;   // plain B (bsrc==MAT)
  const float* feat; int feat_bs;         // (C,NPTS) feature map (gather modes)
  const int*   idx;                       // (NPTS,16) neighbor ids
  float* C; int cs_m, cs_n, cs_b;
  const float* bias; const float* bng; const float* bnb;
  int M, N, K, act;
};

template <int BSRC>
static __device__ inline float slow_load_b(const GemmP& p, int b, int gk, int gn) {
  if (gk >= p.K || gn >= p.N) return 0.f;
  if (BSRC == BSRC_MAT)
    return p.B[b * p.bs_b + gk * p.bs_k + gn * p.bs_n];
  const float* F = p.feat + b * p.feat_bs;
  const int*   I = p.idx  + b * NPTS * KNN;
  int n = gn >> 4, kk = gn & 15;
  int j = I[(n << 4) + kk];
  int C = p.K >> 1;
  if (BSRC == BSRC_EDGE) {
    if (gk < C) return F[gk * NPTS + j] - F[gk * NPTS + n];
    return F[(gk - C) * NPTS + n];
  } else {
    if (gk < C) return F[gk * NPTS + n] - F[gk * NPTS + j];
    return F[(gk - C) * NPTS + j];
  }
}

static __device__ inline void store_tile(const GemmP& p, int b, int tm0, int tn0,
                                         v8f acc, int hi, int mr) {
  int n = tn0 + mr;
  if (n >= p.N) return;
  float* Cb = p.C + b * p.cs_b + n * p.cs_n;
  const float inv = 0.99999499994f; // rsqrt(1+1e-5)
#pragma unroll
  for (int v = 0; v < 8; ++v) {
    int m = tm0 + v + 8 * hi;
    if (m >= p.M) continue;
    float val = acc[v];
    if (p.bias) val += p.bias[m];
    if (p.bng)  val = val * (p.bng[m] * inv) + p.bnb[m];
    if (p.act == ACT_RELU)       val = fmaxf(val, 0.f);
    else if (p.act == ACT_LRELU) val = (val >= 0.f) ? val : 0.2f * val;
    Cb[m * p.cs_m] = val;
  }
}

// 64(M) x 128(N) x 32(K) block tile, 4 waves, each wave 32x64 = 8 WMMAs/K-step.
template <int BSRC>
__global__ __launch_bounds__(128) void gemm_kern(GemmP p) {
  __shared__ unsigned short lA[64][LROW];    // [m][k] bf16
  __shared__ unsigned short lB[128][LROW];   // [n][k] bf16
  const int b   = blockIdx.z;
  const int n0  = blockIdx.x * 128;
  const int m0  = blockIdx.y * 64;
  const int tid = threadIdx.x;
  const int wave = tid >> 5, lane = tid & 31;
  const int hi = lane >> 4, mr = lane & 15;
  const int wm = (wave & 1) * 32, wn = (wave >> 1) * 64;

  const float* Ab = p.A + b * p.as_b;
  const bool mfull = (m0 + 64 <= p.M);
  const bool nfull = (n0 + 128 <= p.N);
  const bool fastA = mfull && (p.as_k == 1);
  const bool fastB = nfull && (BSRC != BSRC_MAT || p.bs_n == 1);

  v8f acc[2][4] = {};

  for (int k0 = 0; k0 < p.K; k0 += 32) {
    bool produced = false;
#if HAVE_ASYNC
    if constexpr (BSRC == BSRC_MAT) {
      // Async DMA path: global f32 tiles -> staging LDS via the async engine
      // (ASYNCcnt), then VALU bf16 conversion out of LDS.  Used for interior
      // blocks on full K-steps (all the heavy head/STN convs).
      __shared__ float stA[64][32];    // [m][k]
      __shared__ float stB[32][128];   // [k][n]
      if (fastA && fastB && (k0 + 32 <= p.K)) {
#pragma unroll
        for (int i = 0; i < 4; ++i) {           // A: 512 b128 chunks
          int chunk = tid + i * 128;
          int m = chunk >> 3, kq = (chunk & 7) * 4;
          __builtin_amdgcn_global_load_async_to_lds_b128(
              (gvp_t)(Ab + (m0 + m) * p.as_m + k0 + kq),
              (svp_t)&stA[m][kq], 0, 0);
        }
        const float* Bb2 = p.B + b * p.bs_b + n0;
#pragma unroll
        for (int i = 0; i < 8; ++i) {           // B: 1024 b128 chunks
          int chunk = tid + i * 128;
          int k = chunk >> 5, nq = (chunk & 31) * 4;
          __builtin_amdgcn_global_load_async_to_lds_b128(
              (gvp_t)(Bb2 + (k0 + k) * p.bs_k + nq),
              (svp_t)&stB[k][nq], 0, 0);
        }
        __builtin_amdgcn_s_wait_asynccnt(0);
        __syncthreads();                        // staging visible to all waves
#pragma unroll
        for (int i = 0; i < 4; ++i) {           // convert A
          int chunk = tid + i * 128;
          int m = chunk >> 3, kq = (chunk & 7) * 4;
          float4 v = *(const float4*)&stA[m][kq];
          *(uint2*)&lA[m][kq] = uint2{pk2(v.x, v.y), pk2(v.z, v.w)};
        }
#pragma unroll
        for (int kq = 0; kq < 32; kq += 4) {    // convert B (column per lane)
          *(uint2*)&lB[tid][kq] = uint2{pk2(stB[kq + 0][tid], stB[kq + 1][tid]),
                                        pk2(stB[kq + 2][tid], stB[kq + 3][tid])};
        }
        produced = true;
      }
    }
#endif
    if (!produced) {
      // ---------------- A tile: thread owns one K-quad for 4 rows ----------------
      if (fastA) {
        const int kq = (tid & 7) * 4;
        const int mb = tid >> 3;                 // 0..15
        if (k0 + kq < p.K) {
#pragma unroll
          for (int i = 0; i < 4; ++i) {
            int m = mb + i * 16;
            const float4 v = *(const float4*)(Ab + (m0 + m) * p.as_m + k0 + kq);
            *(uint2*)&lA[m][kq] = uint2{pk2(v.x, v.y), pk2(v.z, v.w)};
          }
        } else {
#pragma unroll
          for (int i = 0; i < 4; ++i) *(uint2*)&lA[mb + i * 16][kq] = uint2{0u, 0u};
        }
        if (k0 + 32 < p.K)
          __builtin_prefetch(Ab + (m0 + (tid >> 3)) * p.as_m + k0 + 32, 0, 0);
      } else {
#pragma unroll 4
        for (int i = 0; i < 16; ++i) {
          int t = tid + i * 128;
          int m = t >> 5, k = t & 31;
          int gm = m0 + m, gk = k0 + k;
          int cm = gm < p.M ? gm : p.M - 1;
          int ck = gk < p.K ? gk : p.K - 1;
          float v = Ab[cm * p.as_m + ck * p.as_k];
          lA[m][k] = (gm < p.M && gk < p.K) ? bfbits(v) : (unsigned short)0;
        }
      }
      // ---------------- B tile: thread owns one column (n = tid) ----------------
      if (fastB) {
        if (BSRC == BSRC_MAT) {
          const float* Bb = p.B + b * p.bs_b + n0 + tid;
#pragma unroll
          for (int i = 0; i < 8; ++i) {
            int kq = i * 4;
            uint2 w = {0u, 0u};
            if (k0 + kq < p.K) {                // wave-uniform guard (K%4==0)
              float v0 = Bb[(k0 + kq + 0) * p.bs_k];
              float v1 = Bb[(k0 + kq + 1) * p.bs_k];
              float v2 = Bb[(k0 + kq + 2) * p.bs_k];
              float v3 = Bb[(k0 + kq + 3) * p.bs_k];
              w = uint2{pk2(v0, v1), pk2(v2, v3)};
            }
            *(uint2*)&lB[tid][kq] = w;
          }
          if (k0 + 32 < p.K)
            __builtin_prefetch(Bb + (k0 + 32) * p.bs_k, 0, 0);
        } else {
          const float* F = p.feat + b * p.feat_bs;
          const int col = n0 + tid;
          const int pt = col >> 4, kk = col & 15;
          const int j = p.idx[b * NPTS * KNN + (pt << 4) + kk];
          const int C = p.K >> 1;               // multiple of 4
#pragma unroll
          for (int i = 0; i < 8; ++i) {
            int gk = k0 + i * 4;
            uint2 w = {0u, 0u};
            if (gk < p.K) {                     // quad never straddles K or C
              float v[4];
              if (BSRC == BSRC_EDGE) {
                if (gk < C) {
#pragma unroll
                  for (int q = 0; q < 4; ++q)
                    v[q] = F[(gk + q) * NPTS + j] - F[(gk + q) * NPTS + pt];
                } else {
#pragma unroll
                  for (int q = 0; q < 4; ++q) v[q] = F[(gk - C + q) * NPTS + pt];
                }
              } else {
                if (gk < C) {
#pragma unroll
                  for (int q = 0; q < 4; ++q)
                    v[q] = F[(gk + q) * NPTS + pt] - F[(gk + q) * NPTS + j];
                } else {
#pragma unroll
                  for (int q = 0; q < 4; ++q) v[q] = F[(gk - C + q) * NPTS + j];
                }
              }
              w = uint2{pk2(v[0], v[1]), pk2(v[2], v[3])};
            }
            *(uint2*)&lB[tid][i * 4] = w;
          }
        }
      } else {
#pragma unroll 4
        for (int i = 0; i < 32; ++i) {
          int t = tid + i * 128;
          int n = t >> 5, k = t & 31;
          lB[n][k] = bfbits(slow_load_b<BSRC>(p, b, k0 + k, n0 + n));
        }
      }
    }
    __syncthreads();

    const unsigned* rA0 = (const unsigned*)(&lA[wm + mr][0]);
    const unsigned* rA1 = (const unsigned*)(&lA[wm + 16 + mr][0]);
    FragB fa0, fa1;
#pragma unroll
    for (int j = 0; j < 8; ++j) {
      int da = (j < 4) ? (4 * hi + j) : (4 + 4 * hi + j);  // A 16x32 bf16 layout
      fa0.u[j] = rA0[da];
      fa1.u[j] = rA1[da];
    }
#pragma unroll
    for (int tj = 0; tj < 4; ++tj) {
      const unsigned* rB = (const unsigned*)(&lB[wn + 16 * tj + mr][0]);
      FragB fb;
#pragma unroll
      for (int j = 0; j < 8; ++j) fb.u[j] = rB[(j < 4) ? (4 * hi + j) : (4 + 4 * hi + j)];  // B 32x16 bf16 layout
      acc[0][tj] = __builtin_amdgcn_wmma_f32_16x16x32_bf16(false, fa0.v, false, fb.v, (short)0, acc[0][tj], false, false);
      acc[1][tj] = __builtin_amdgcn_wmma_f32_16x16x32_bf16(false, fa1.v, false, fb.v, (short)0, acc[1][tj], false, false);
    }
    __syncthreads();
  }
#pragma unroll
  for (int ti = 0; ti < 2; ++ti)
#pragma unroll
    for (int tj = 0; tj < 4; ++tj)
      store_tile(p, b, m0 + wm + 16 * ti, n0 + wn + 16 * tj, acc[ti][tj], hi, mr);
}

// ---- kNN: fused distance tiles (f32 WMMA) + register-resident top-16 ----
__global__ __launch_bounds__(32) void knn_topk(const float* __restrict__ feat,
                                               int feat_bs,
                                               const float* __restrict__ xx,
                                               int* __restrict__ idxOut, int C) {
  __shared__ float sA[128 * 16];     // [c][m] row block coords (C<=128)
  __shared__ float sD[16][16];
  __shared__ float sXX[16];
  const int b = blockIdx.y, m0 = blockIdx.x * 16;
  const int lane = threadIdx.x, hi = lane >> 4, mr = lane & 15;
  const float* F = feat + b * feat_bs;
  const float* X = xx + b * NPTS;

  for (int t = lane; t < C * 16; t += 32)
    sA[t] = F[(t >> 4) * NPTS + m0 + (t & 15)];
  __syncthreads();

  float tv[16]; int ti[16];
#pragma unroll
  for (int j = 0; j < 16; ++j) { tv[j] = -3.4e38f; ti[j] = 0; }
  float curmin = -3.4e38f; int minpos = 0;
  float xr = (lane < 16) ? X[m0 + lane] : 0.f;

  const int cA = hi ? 2 : 0, cB = hi ? 3 : 1;
  for (int n0 = 0; n0 < NPTS; n0 += 16) {
#if __has_builtin(__builtin_amdgcn_wmma_f32_16x16x4_f32)
    v8f acc = {};
    for (int c0 = 0; c0 < C; c0 += 4) {
      v2f a, bb;
      a.x  = sA[(c0 + cA) * 16 + mr];
      a.y  = sA[(c0 + cB) * 16 + mr];
      bb.x = F[(c0 + cA) * NPTS + n0 + mr];
      bb.y = F[(c0 + cB) * NPTS + n0 + mr];
      acc = __builtin_amdgcn_wmma_f32_16x16x4_f32(false, a, false, bb, (short)0, acc, false, false);
    }
#pragma unroll
    for (int v = 0; v < 8; ++v) sD[v + 8 * hi][mr] = acc[v];
#else
#pragma unroll
    for (int v = 0; v < 8; ++v) {
      int m = v + 8 * hi;
      float s = 0.f;
      for (int c = 0; c < C; ++c) s += sA[c * 16 + m] * F[c * NPTS + n0 + mr];
      sD[m][mr] = s;
    }
#endif
    if (lane < 16) sXX[lane] = X[n0 + lane];
    __syncthreads();
    if (lane < 16) {
#pragma unroll
      for (int j = 0; j < 16; ++j) {
        float nd = 2.f * sD[lane][j] - xr - sXX[j];
        if (nd > curmin) {
#pragma unroll
          for (int q = 0; q < 16; ++q)
            if (q == minpos) { tv[q] = nd; ti[q] = n0 + j; }
          curmin = tv[0]; minpos = 0;
#pragma unroll
          for (int q = 1; q < 16; ++q)
            if (tv[q] < curmin) { curmin = tv[q]; minpos = q; }
        }
      }
    }
    __syncthreads();
  }
  if (lane < 16) {
#pragma unroll
    for (int j = 0; j < 16; ++j)
      idxOut[(b * NPTS + m0 + lane) * KNN + j] = ti[j];
  }
}

// ---- small helper kernels ----
__global__ void sumsq_kernel(const float* feat, int feat_bs, int C, float* xx) {
  int t = blockIdx.x * blockDim.x + threadIdx.x;
  if (t >= NB * NPTS) return;
  int b = t / NPTS, n = t % NPTS;
  const float* F = feat + b * feat_bs + n;
  float s = 0.f;
  for (int c = 0; c < C; ++c) { float v = F[c * NPTS]; s += v * v; }
  xx[t] = s;
}

__global__ __launch_bounds__(256) void rowmax_kernel(const float* in, int in_bs,
                                                     float* out, int rows) {
  int bc = blockIdx.x;
  int b = bc / rows, c = bc % rows;
  const float* R = in + b * in_bs + c * NPTS;
  float m = -3.4e38f;
  for (int n = threadIdx.x; n < NPTS; n += 256) m = fmaxf(m, R[n]);
  __shared__ float sm[256];
  sm[threadIdx.x] = m;
  __syncthreads();
  for (int s = 128; s > 0; s >>= 1) {
    if (threadIdx.x < s) sm[threadIdx.x] = fmaxf(sm[threadIdx.x], sm[threadIdx.x + s]);
    __syncthreads();
  }
  if (threadIdx.x == 0) out[bc] = sm[0];
}

__global__ void stn_apply_kernel(const float* x, int x_bs, const float* T, float* out) {
  int t = blockIdx.x * blockDim.x + threadIdx.x;
  if (t >= NB * 12 * NPTS) return;
  int n = t % NPTS, d = (t / NPTS) % 12, b = t / (12 * NPTS);
  const float* X = x + b * x_bs + n;
  const float* Tm = T + b * 144;
  float s = 0.f;
#pragma unroll
  for (int c = 0; c < 12; ++c)
    s += X[c * NPTS] * (Tm[c * 12 + d] + (c == d ? 1.f : 0.f));
  out[t] = s;
}

__global__ void maxk_kernel(const float* in, int F, float* out, int out_bs) {
  int t = blockIdx.x * blockDim.x + threadIdx.x;
  if (t >= NB * F * NPTS) return;
  int n = t % NPTS, f = (t / NPTS) % F, b = t / (F * NPTS);
  const float4* P = (const float4*)(in + ((b * F + f) * NPTS + n) * (long long)KNN);
  float m = -3.4e38f;
#pragma unroll
  for (int j = 0; j < 4; ++j) {
    float4 v = P[j];
    m = fmaxf(m, fmaxf(fmaxf(v.x, v.y), fmaxf(v.z, v.w)));
  }
  out[b * out_bs + f * NPTS + n] = m;
}

__global__ void att_reduce_kernel(const float* e, const float* feat, int F,
                                  float* out, int out_bs) {
  int t = blockIdx.x * blockDim.x + threadIdx.x;
  if (t >= NB * F * NPTS) return;
  int n = t % NPTS, f = (t / NPTS) % F, b = t / (F * NPTS);
  long long base = ((b * F + f) * NPTS + n) * (long long)KNN;
  float ev[16], mx = -3.4e38f;
#pragma unroll
  for (int j = 0; j < 16; ++j) { ev[j] = e[base + j]; mx = fmaxf(mx, ev[j]); }
  float s = 0.f;
#pragma unroll
  for (int j = 0; j < 16; ++j) { ev[j] = __expf(ev[j] - mx); s += ev[j]; }
  float r = 0.f;
#pragma unroll
  for (int j = 0; j < 16; ++j) r += ev[j] * feat[base + j];
  out[b * out_bs + f * NPTS + n] = r / s;
}

__global__ void chmean_kernel(const float* in, int in_bs, int C, float* out) {
  int t = blockIdx.x * blockDim.x + threadIdx.x;
  if (t >= NB * NPTS) return;
  int b = t / NPTS, n = t % NPTS;
  const float* P = in + b * in_bs + n;
  float s = 0.f;
  for (int c = 0; c < C; ++c) s += P[c * NPTS];
  out[t] = s / (float)C;
}

__global__ void hscale_kernel(float* H, const float* ac, const float* an) {
  int t = blockIdx.x * blockDim.x + threadIdx.x;
  if (t >= NB * 1024 * NPTS) return;
  int n = t % NPTS, c = (t / NPTS) % 1024, b = t / (1024 * NPTS);
  float a = ac[b * NPTS + n], bb = an[b * NPTS + n];
  float s = a + bb;
  H[t] *= (c < 512 ? a : bb) / s;
}

__global__ void mul_kernel(float* H, const float* W, int total) {
  int t = blockIdx.x * blockDim.x + threadIdx.x;
  if (t < total) H[t] *= W[t];
}

// ---- host orchestration ----
static inline int nblk(long long n) { return (int)((n + 255) / 256); }

static void launch_gemm(hipStream_t s,
    const float* A, long long as_m, long long as_k, long long as_b,
    const float* B, long long bs_k, long long bs_n, long long bs_b,
    const float* feat, long long feat_bs, const int* idx, int bsrc,
    float* C, long long cs_m, long long cs_n, long long cs_b,
    const float* bias, const float* bng, const float* bnb,
    int M, int N, int K, int act, int batches) {
  GemmP p;
  p.A = A; p.as_m = (int)as_m; p.as_k = (int)as_k; p.as_b = (int)as_b;
  p.B = B; p.bs_k = (int)bs_k; p.bs_n = (int)bs_n; p.bs_b = (int)bs_b;
  p.feat = feat; p.feat_bs = (int)feat_bs; p.idx = idx;
  p.C = C; p.cs_m = (int)cs_m; p.cs_n = (int)cs_n; p.cs_b = (int)cs_b;
  p.bias = bias; p.bng = bng; p.bnb = bnb;
  p.M = M; p.N = N; p.K = K; p.act = act;
  dim3 g((N + 127) / 128, (M + 63) / 64, batches);
  if (bsrc == BSRC_MAT)       gemm_kern<BSRC_MAT ><<<g, 128, 0, s>>>(p);
  else if (bsrc == BSRC_EDGE) gemm_kern<BSRC_EDGE><<<g, 128, 0, s>>>(p);
  else                        gemm_kern<BSRC_ATT ><<<g, 128, 0, s>>>(p);
}

extern "C" void kernel_launch(void* const* d_in, const int* in_sizes, int n_in,
                              void* d_out, int out_size, void* d_ws, size_t ws_size,
                              hipStream_t stream) {
  (void)in_sizes; (void)n_in; (void)out_size; (void)ws_size;
  const float* x = (const float*)d_in[0];
  auto P = [&](int i) { return (const float*)d_in[i]; };
  float* out = (float*)d_out;

  float* ws = (float*)d_ws;
  size_t o = 0;
  auto alloc = [&](size_t n) { float* p = ws + o; o += n; return p; };
  float* XX    = alloc((size_t)NB * NPTS);
  int*   IDX   = (int*)alloc((size_t)NB * NPTS * KNN);
  float* COORT = alloc((size_t)NB * 12 * NPTS);
  float* NORT  = alloc((size_t)NB * 12 * NPTS);
  float* CC    = alloc((size_t)NB * 448 * NPTS);   // [coor1|coor2|coor3]
  float* NC    = alloc((size_t)NB * 448 * NPTS);   // [nor1 |nor2 |nor3 ]
  float* H     = alloc((size_t)NB * 1024 * NPTS);  // [coorf|norf]
  float* WB    = alloc((size_t)NB * 1024 * NPTS);  // stn h1024 / fa weights
  float* T512  = alloc((size_t)NB * 512 * NPTS);
  float* T256  = alloc((size_t)NB * 256 * NPTS);
  float* T128B = alloc((size_t)NB * 128 * NPTS);
  float* AVGC  = alloc((size_t)NB * NPTS);
  float* AVGN  = alloc((size_t)NB * NPTS);
  float* V1024 = alloc(2048);
  float* V512  = alloc(1024);
  float* V256  = alloc(512);
  float* TMAT  = alloc(288);
  float* BIG1  = alloc((size_t)NB * 256 * NPTS * KNN);
  float* BIG2  = alloc((size_t)NB * 256 * NPTS * KNN);

  const long long XBS = 24LL * NPTS;

  // ===== STN for coor (params 1..22) and nor (23..44) =====
  for (int si = 0; si < 2; ++si) {
    const float* xin = x + (si ? 12 * NPTS : 0);
    int pb = si ? 23 : 1;
    launch_gemm(stream, P(pb + 0), 12, 1, 0, xin, NPTS, 1, XBS,
                nullptr, 0, nullptr, BSRC_MAT,
                T512, NPTS, 1, 64LL * NPTS,
                P(pb + 1), P(pb + 12), P(pb + 13), 64, NPTS, 12, ACT_RELU, NB);
    launch_gemm(stream, P(pb + 2), 64, 1, 0, T512, NPTS, 1, 64LL * NPTS,
                nullptr, 0, nullptr, BSRC_MAT,
                T128B, NPTS, 1, 128LL * NPTS,
                P(pb + 3), P(pb + 14), P(pb + 15), 128, NPTS, 64, ACT_RELU, NB);
    launch_gemm(stream, P(pb + 4), 128, 1, 0, T128B, NPTS, 1, 128LL * NPTS,
                nullptr, 0, nullptr, BSRC_MAT,
                WB, NPTS, 1, 1024LL * NPTS,
                P(pb + 5), P(pb + 16), P(pb + 17), 1024, NPTS, 128, ACT_RELU, NB);
    rowmax_kernel<<<NB * 1024, 256, 0, stream>>>(WB, 1024 * NPTS, V1024, 1024);
    launch_gemm(stream, P(pb + 6), 1024, 1, 0, V1024, 1, 1024, 0,
                nullptr, 0, nullptr, BSRC_MAT,
                V512, 1, 512, 0,
                P(pb + 7), P(pb + 18), P(pb + 19), 512, NB, 1024, ACT_RELU, 1);
    launch_gemm(stream, P(pb + 8), 512, 1, 0, V512, 1, 512, 0,
                nullptr, 0, nullptr, BSRC_MAT,
                V256, 1, 256, 0,
                P(pb + 9), P(pb + 20), P(pb + 21), 256, NB, 512, ACT_RELU, 1);
    launch_gemm(stream, P(pb + 10), 256, 1, 0, V256, 1, 256, 0,
                nullptr, 0, nullptr, BSRC_MAT,
                TMAT, 1, 144, 0,
                P(pb + 11), nullptr, nullptr, 144, NB, 256, ACT_NONE, 1);
    stn_apply_kernel<<<nblk(NB * 12 * NPTS), 256, 0, stream>>>(
        xin, 24 * NPTS, TMAT, si ? NORT : COORT);
  }

  // ===== graph layers =====
  struct FeatRef { const float* p; int bs; int C; };
  auto layer = [&](FeatRef cF, FeatRef nF, int wc, int wn, int wa, int Fo,
                   float* coorOut, float* norOut) {
    sumsq_kernel<<<nblk(NB * NPTS), 256, 0, stream>>>(cF.p, cF.bs, cF.C, XX);
    knn_topk<<<dim3(NPTS / 16, NB), 32, 0, stream>>>(cF.p, cF.bs, XX, IDX, cF.C);
    const long long N2 = 16LL * NPTS;
    launch_gemm(stream, P(wn), 2 * nF.C, 1, 0, nullptr, 0, 0, 0,
                nF.p, nF.bs, IDX, BSRC_EDGE,
                BIG1, N2, 1, (long long)Fo * N2,
                nullptr, P(wn + 1), P(wn + 2), Fo, (int)N2, 2 * nF.C, ACT_LRELU, NB);
    maxk_kernel<<<nblk((long long)NB * Fo * NPTS), 256, 0, stream>>>(BIG1, Fo, norOut, 448 * NPTS);
    launch_gemm(stream, P(wc), 2 * cF.C, 1, 0, nullptr, 0, 0, 0,
                cF.p, cF.bs, IDX, BSRC_EDGE,
                BIG1, N2, 1, (long long)Fo * N2,
                nullptr, P(wc + 1), P(wc + 2), Fo, (int)N2, 2 * cF.C, ACT_LRELU, NB);
    launch_gemm(stream, P(wa), 2 * cF.C, 1, 0, nullptr, 0, 0, 0,
                cF.p, cF.bs, IDX, BSRC_ATT,
                BIG2, N2, 1, (long long)Fo * N2,
                nullptr, P(wa + 1), P(wa + 2), Fo, (int)N2, 2 * cF.C, ACT_LRELU, NB);
    att_reduce_kernel<<<nblk((long long)NB * Fo * NPTS), 256, 0, stream>>>(
        BIG2, BIG1, Fo, coorOut, 448 * NPTS);
  };

  layer({COORT, 12 * NPTS, 12}, {NORT, 12 * NPTS, 12}, 54, 66, 45, 64, CC, NC);
  layer({CC, 448 * NPTS, 64}, {NC, 448 * NPTS, 64}, 57, 69, 48, 128,
        CC + 64 * NPTS, NC + 64 * NPTS);
  layer({CC + 64 * NPTS, 448 * NPTS, 128}, {NC + 64 * NPTS, 448 * NPTS, 128},
        60, 72, 51, 256, CC + 192 * NPTS, NC + 192 * NPTS);

  // ===== fusion head =====
  launch_gemm(stream, P(63), 448, 1, 0, CC, NPTS, 1, 448LL * NPTS,
              nullptr, 0, nullptr, BSRC_MAT,
              H, NPTS, 1, 1024LL * NPTS,
              nullptr, P(64), P(65), 512, NPTS, 448, ACT_LRELU, NB);
  launch_gemm(stream, P(75), 448, 1, 0, NC, NPTS, 1, 448LL * NPTS,
              nullptr, 0, nullptr, BSRC_MAT,
              H + 512 * NPTS, NPTS, 1, 1024LL * NPTS,
              nullptr, P(76), P(77), 512, NPTS, 448, ACT_LRELU, NB);
  chmean_kernel<<<nblk(NB * NPTS), 256, 0, stream>>>(H, 1024 * NPTS, 512, AVGC);
  chmean_kernel<<<nblk(NB * NPTS), 256, 0, stream>>>(H + 512 * NPTS, 1024 * NPTS, 512, AVGN);
  hscale_kernel<<<nblk((long long)NB * 1024 * NPTS), 256, 0, stream>>>(H, AVGC, AVGN);
  launch_gemm(stream, P(78), 1024, 1, 0, H, NPTS, 1, 1024LL * NPTS,
              nullptr, 0, nullptr, BSRC_MAT,
              WB, NPTS, 1, 1024LL * NPTS,
              nullptr, P(79), P(80), 1024, NPTS, 1024, ACT_LRELU, NB);
  mul_kernel<<<nblk((long long)NB * 1024 * NPTS), 256, 0, stream>>>(H, WB, NB * 1024 * NPTS);
  launch_gemm(stream, P(81), 1024, 1, 0, H, NPTS, 1, 1024LL * NPTS,
              nullptr, 0, nullptr, BSRC_MAT,
              T512, NPTS, 1, 512LL * NPTS,
              nullptr, P(82), P(83), 512, NPTS, 1024, ACT_LRELU, NB);
  launch_gemm(stream, P(84), 512, 1, 0, T512, NPTS, 1, 512LL * NPTS,
              nullptr, 0, nullptr, BSRC_MAT,
              T256, NPTS, 1, 256LL * NPTS,
              nullptr, P(85), P(86), 256, NPTS, 512, ACT_LRELU, NB);
  launch_gemm(stream, P(87), 256, 1, 0, T256, NPTS, 1, 256LL * NPTS,
              nullptr, 0, nullptr, BSRC_MAT,
              T128B, NPTS, 1, 128LL * NPTS,
              nullptr, P(88), P(89), 128, NPTS, 256, ACT_LRELU, NB);
  launch_gemm(stream, P(90), 128, 1, 0, T128B, NPTS, 1, 128LL * NPTS,
              nullptr, 0, nullptr, BSRC_MAT,
              out, 1, 17, 17LL * NPTS,
              nullptr, nullptr, nullptr, 17, NPTS, 128, ACT_NONE, NB);
}
